// LifNetwork_11622181503296
// MI455X (gfx1250) — compile-verified, matrix operand order checked by reading, not run on hardware
//
#include <hip/hip_runtime.h>
#include <hip/hip_bf16.h>

// ---------------- problem constants (mirror reference) ----------------
#define N_BATCH   64
#define N_NEURON  8192
#define N_STEPS   100
#define N_STEADY  50
#define N_WINDOW  50

#define DT_F           0.1f
#define EXP_DT_TAU     0.99501247919268232f   // exp(-0.1/20)
#define EXP_DT_TAU_SYN 0.95122942450071402f   // exp(-0.1/2)
#define DT_TAU_SYN     0.05f
#define V_THRESH       1.0f
#define V_REST         0.0f

#define STATE_ELEMS (N_BATCH * N_NEURON)      // 524288
#define KSPLIT      4
#define KQ_LEN      (N_NEURON / KSPLIT)       // 2048

// LDS tiling for the WMMA kernel
#define KCHUNK   64                            // K elems staged per chunk
#define ROWB     144                           // padded LDS row pitch (128B data + 16B pad)
#define WTILE_B  (128 * ROWB)                  // 18432 B : 128 W rows
#define STILE_B  (64 * ROWB)                   //  9216 B : 64 S rows
#define BUF_B    (WTILE_B + STILE_B)           // 27648 B per buffer (x2 = 54 KiB)

typedef __attribute__((ext_vector_type(16))) __bf16 v16bf;
typedef __attribute__((ext_vector_type(8)))  __bf16 v8bf;
typedef __attribute__((ext_vector_type(8)))  float  v8f;
typedef __attribute__((ext_vector_type(4)))  unsigned int u32x4;
typedef __attribute__((ext_vector_type(4)))  int i32x4;
typedef __attribute__((ext_vector_type(8)))  int i32x8;

#define WMMA_BF16(A, B, C) \
    __builtin_amdgcn_wmma_f32_16x16x32_bf16(false, (A), false, (B), (short)0, (C), false, false)

// ---- TDM: DMA a 16-row x 64-col bf16 tile (row stride 8192) into LDS --------
// D# per CDNA5 ISA. LDS rows padded +4 DWORDs every
// 32 DWORDs -> 144 B pitch (matches ROWB).
__device__ __forceinline__ void tdm_load_16x64(unsigned lds_off, const void* gsrc) {
    const unsigned long long ga = (unsigned long long)(uintptr_t)gsrc;
    u32x4 g0;
    g0[0] = 1u;                                                // count=1 user D#
    g0[1] = lds_off;                                           // LDS dest (bytes)
    g0[2] = (unsigned)ga;                                      // global_addr[31:0]
    g0[3] = (unsigned)((ga >> 32) & 0x01FFFFFFu) | 0x80000000u;// addr[56:32] | type=2
    i32x8 g1;
    g1[0] = (1 << 16) | (1 << 20) | (4 << 22) | (3 << 25);     // 2B elems, pad 4dw/32dw
    g1[1] = (int)(KCHUNK << 16);                               // tensor_dim0 = 64
    g1[2] = (int)(16 << 16);                                   // tensor_dim1 = 16
    g1[3] = (int)(KCHUNK << 16);                               // tile_dim0 = 64
    g1[4] = 16;                                                // tile_dim1 = 16
    g1[5] = N_NEURON;                                          // tensor_dim0_stride
    g1[6] = 0;
    g1[7] = 0;
    const i32x4 z4 = {0, 0, 0, 0};
#if __clang_major__ >= 23
    const i32x8 z8 = {0, 0, 0, 0, 0, 0, 0, 0};
    __builtin_amdgcn_tensor_load_to_lds(g0, g1, z4, z4, z8, 0);
#else
    __builtin_amdgcn_tensor_load_to_lds(g0, g1, z4, z4, 0);
#endif
}

// assemble a 32-B WMMA fragment from two 16-B LDS loads (16-B aligned pitch)
__device__ __forceinline__ v16bf lds_frag(const char* p) {
    const v8bf lo = *(const v8bf*)(p);
    const v8bf hi = *(const v8bf*)(p + 16);
    return __builtin_shufflevector(lo, hi, 0, 1, 2, 3, 4, 5, 6, 7,
                                   8, 9, 10, 11, 12, 13, 14, 15);
}

// ---------------- 1) W (fp32, row-major [n][k]) -> bf16 ----------------
__global__ void k_w_to_bf16(const float* __restrict__ W, __bf16* __restrict__ Wb) {
    const size_t i = (size_t)blockIdx.x * blockDim.x + threadIdx.x;   // 8 elems/thread
    const size_t base = i * 8;
    const float4 a = ((const float4*)W)[i * 2 + 0];
    const float4 b = ((const float4*)W)[i * 2 + 1];
    v8bf o;
    o[0] = (__bf16)a.x; o[1] = (__bf16)a.y; o[2] = (__bf16)a.z; o[3] = (__bf16)a.w;
    o[4] = (__bf16)b.x; o[5] = (__bf16)b.y; o[6] = (__bf16)b.z; o[7] = (__bf16)b.w;
    *(v8bf*)(Wb + base) = o;
}

// ---------------- 2) init state in workspace ----------------
__global__ void k_init(const float* __restrict__ volt0,
                       float* __restrict__ volt, float* __restrict__ rec,
                       float* __restrict__ sc, int* __restrict__ isi,
                       int* __restrict__ st, __bf16* __restrict__ sbuf,
                       float* __restrict__ part) {
    const int i = blockIdx.x * blockDim.x + threadIdx.x;
    if (i < STATE_ELEMS) {
        volt[i] = volt0[i];
        rec[i]  = 0.0f;
        sc[i]   = 0.0f;
        isi[i]  = 0;
        st[i]   = 0;
        sbuf[i] = (__bf16)0.0f;
        part[i]                   = 0.0f;
        part[STATE_ELEMS + i]     = 0.0f;
        part[2 * STATE_ELEMS + i] = 0.0f;
        part[3 * STATE_ELEMS + i] = 0.0f;
    }
}

// ---------------- 3a) fused combine + LIF step, emits bf16 spike matrix -------
__global__ void k_step(float* __restrict__ volt, float* __restrict__ rec,
                       const float* __restrict__ ff, __bf16* __restrict__ sbuf,
                       int* __restrict__ isi, int* __restrict__ st,
                       float* __restrict__ sc, const float* __restrict__ part,
                       int step) {
    const int i = blockIdx.x * blockDim.x + threadIdx.x;
    if (i >= STATE_ELEMS) return;
    const float psum = part[i] + part[STATE_ELEMS + i]
                     + part[2 * STATE_ELEMS + i] + part[3 * STATE_ELEMS + i];
    const float r = rec[i] * EXP_DT_TAU_SYN + DT_TAU_SYN * psum;
    rec[i] = r;
    float v = volt[i] * EXP_DT_TAU + DT_F * (ff[i] + r);
    const bool sp = (v >= V_THRESH);
    volt[i] = sp ? V_REST : v;
    if (sp) {
        isi[i] = step - st[i];
        st[i]  = step;
        if (step >= N_STEADY) sc[i] += 1.0f;
    }
    sbuf[i] = sp ? (__bf16)1.0f : (__bf16)0.0f;
}

// ---------------- 3b) partial[kq] = s @ W^T over one K-quarter ----------------
// 8 waves; wave w owns 16-neuron N-tile + all 4 M-tiles. TDM double-buffers
// W (128x64) and S (64x64) tiles into LDS (each wave DMAs its own 16 rows;
// waves 0-3 also DMA 16 spike rows). WMMAs read short-latency ds_load_b128
// fragments while the TDM streams the next chunk (TENSORcnt + barrier sync).
__global__ __launch_bounds__(256)
void k_rec_wmma(const __bf16* __restrict__ S, const __bf16* __restrict__ Wb,
                float* __restrict__ part) {
    __shared__ __align__(16) char lds[2][BUF_B];

    const int lane = threadIdx.x & 31;
    const int wave = threadIdx.x >> 5;                 // 0..7
    const int n0   = blockIdx.x * 128 + wave * 16;     // 16-neuron N-tile
    const int kq   = blockIdx.y;                       // K-quarter
    const size_t kbase = (size_t)kq * KQ_LEN;

    // per-wave TDM sources (16 rows each, stride N_NEURON)
    const __bf16* wsrc = Wb + (size_t)n0 * N_NEURON + kbase;
    const __bf16* ssrc = S + (size_t)(wave & 3) * 16 * N_NEURON + kbase;

    const unsigned lds0 = (unsigned)(uintptr_t)&lds[0][0];
    const unsigned wsub = (unsigned)(wave * 16 * ROWB);

    // fragment base addresses in LDS (row r of a tile at r*ROWB)
    const int half2 = ((lane >> 4) << 4) * 2;          // +0 / +32 B (K halves)
    const char* wrow = &lds[0][0] + (wave * 16 + (lane & 15)) * ROWB + half2;
    const char* srow = &lds[0][0] + WTILE_B + (lane & 15) * ROWB + half2;

    // prologue: DMA chunk 0 into buffer 0
    tdm_load_16x64(lds0 + wsub, wsrc);
    if (wave < 4) tdm_load_16x64(lds0 + WTILE_B + wsub, ssrc);

    v8f acc0 = {}, acc1 = {}, acc2 = {}, acc3 = {};
    const int NCH = KQ_LEN / KCHUNK;                   // 32 chunks
#pragma unroll 1
    for (int c = 0; c < NCH; ++c) {
        __builtin_amdgcn_s_wait_tensorcnt(0);          // own TDM(s) for chunk c done
        __syncthreads();                               // => all waves' tiles ready
        if (c + 1 < NCH) {                             // DMA next chunk, other buffer
            const unsigned bb = lds0 + (unsigned)(((c + 1) & 1) * BUF_B);
            tdm_load_16x64(bb + wsub, wsrc + (size_t)(c + 1) * KCHUNK);
            if (wave < 4)
                tdm_load_16x64(bb + WTILE_B + wsub, ssrc + (size_t)(c + 1) * KCHUNK);
        }
        const char* wb_ = wrow + (c & 1) * BUF_B;
        const char* sb_ = srow + (c & 1) * BUF_B;
#pragma unroll
        for (int kk = 0; kk < KCHUNK; kk += 32) {
            const int kb = kk * 2;                     // byte offset within row
            const v16bf b  = lds_frag(wb_ + kb);
            const v16bf a0 = lds_frag(sb_ + kb);
            const v16bf a1 = lds_frag(sb_ + 16 * ROWB + kb);
            const v16bf a2 = lds_frag(sb_ + 32 * ROWB + kb);
            const v16bf a3 = lds_frag(sb_ + 48 * ROWB + kb);
            acc0 = WMMA_BF16(a0, b, acc0);
            acc1 = WMMA_BF16(a1, b, acc1);
            acc2 = WMMA_BF16(a2, b, acc2);
            acc3 = WMMA_BF16(a3, b, acc3);
        }
    }

    // C/D layout: VGPR j -> M = j + 8*(lane>=16), N = lane&15
    float* P = part + (size_t)kq * STATE_ELEMS;
    const int col  = n0 + (lane & 15);
    const int rsub = (lane >> 4) << 3;
#pragma unroll
    for (int j = 0; j < 8; ++j) {
        const int row = rsub + j;
        P[(size_t)(row)      * N_NEURON + col] = acc0[j];
        P[(size_t)(row + 16) * N_NEURON + col] = acc1[j];
        P[(size_t)(row + 32) * N_NEURON + col] = acc2[j];
        P[(size_t)(row + 48) * N_NEURON + col] = acc3[j];
    }
}

// ---------------- 4) finalize: fold in step-99 partials, pack outputs ---------
__global__ void k_final(const float* __restrict__ volt, const float* __restrict__ rec,
                        const float* __restrict__ sc, const float* __restrict__ part,
                        float* __restrict__ out) {
    const int i = blockIdx.x * blockDim.x + threadIdx.x;
    if (i < STATE_ELEMS) {
        const float psum = part[i] + part[STATE_ELEMS + i]
                         + part[2 * STATE_ELEMS + i] + part[3 * STATE_ELEMS + i];
        out[i]                   = volt[i];
        out[STATE_ELEMS + i]     = rec[i] * EXP_DT_TAU_SYN + DT_TAU_SYN * psum;
        out[2 * STATE_ELEMS + i] = sc[i] * (1.0f / N_WINDOW);
    }
}

// ---------------- host: launch sequence (graph-capture safe) ----------------
extern "C" void kernel_launch(void* const* d_in, const int* in_sizes, int n_in,
                              void* d_out, int out_size, void* d_ws, size_t ws_size,
                              hipStream_t stream) {
    const float* volt0 = (const float*)d_in[0];
    const float* W     = (const float*)d_in[1];
    const float* ff    = (const float*)d_in[2];
    float* out = (float*)d_out;

    // workspace layout (bytes)
    char* ws = (char*)d_ws;
    __bf16* Wb   = (__bf16*)(ws);                                   // 128 MiB
    float*  volt = (float*)(ws + 134217728);                        // 2 MiB
    float*  rec  = (float*)(ws + 134217728 + 1 * 2097152);          // 2 MiB
    float*  sc   = (float*)(ws + 134217728 + 2 * 2097152);          // 2 MiB
    int*    isi  = (int*)  (ws + 134217728 + 3 * 2097152);          // 2 MiB
    int*    st   = (int*)  (ws + 134217728 + 4 * 2097152);          // 2 MiB
    __bf16* sbuf = (__bf16*)(ws + 134217728 + 5 * 2097152);         // 1 MiB
    float*  part = (float*)(ws + 134217728 + 5 * 2097152 + 1048576);// 4 x 2 MiB

    // 1) convert W to bf16 (8 elems/thread): 64M elems -> 32768 blocks
    k_w_to_bf16<<<dim3(32768), dim3(256), 0, stream>>>(W, Wb);

    // 2) init state (+ zero partials so step 0 sees rec_{-1} = 0)
    k_init<<<dim3(STATE_ELEMS / 256), dim3(256), 0, stream>>>(
        volt0, volt, rec, sc, isi, st, sbuf, part);

    // 3) time loop: fused (combine + LIF) then K-split TDM+WMMA partial GEMM
    for (int step = 0; step < N_STEPS; ++step) {
        k_step<<<dim3(STATE_ELEMS / 256), dim3(256), 0, stream>>>(
            volt, rec, ff, sbuf, isi, st, sc, part, step);
        k_rec_wmma<<<dim3(N_NEURON / 128, KSPLIT), dim3(256), 0, stream>>>(
            sbuf, Wb, part);
    }

    // 4) finalize: rec_99 = e*rec_98 + ds*sum(partials of s_99); pack outputs
    k_final<<<dim3(STATE_ELEMS / 256), dim3(256), 0, stream>>>(
        volt, rec, sc, part, out);
}
